// CrossAttentionModuleWMask_13022340841955
// MI455X (gfx1250) — matrix operand order repaired and checked by reading, not compiled, not run on hardware
//
#include <hip/hip_runtime.h>
#include <hip/hip_bf16.h>

typedef __attribute__((ext_vector_type(16))) __bf16 v16bf;
typedef __attribute__((ext_vector_type(8)))  float  v8f;
typedef __attribute__((ext_vector_type(4)))  unsigned int u32x4;
typedef __attribute__((ext_vector_type(8)))  int  i32x8;
typedef __attribute__((ext_vector_type(4)))  int  i32x4;

union BfVec {
    v16bf v;
    uint4  q[2];
    __bf16 h[16];
};

#define D_DIM   1024
#define N_IMG   257
#define L_LM    9
#define HEADS   16
#define BT_DIM  128
#define ROWS_Q  (BT_DIM * N_IMG)   // 32896 = 1028*32
#define ROWS_KV (BT_DIM * L_LM)    // 1152  = 36*32

// LDS A-slab row stride: 512 data dwords + 2x4 pad dwords (TDM pad_interval=256dw, pad_amount=4dw)
#define LDS_ROW_DW 520

// ---------------------------------------------------------------- converts
__global__ __launch_bounds__(256) void f32_to_bf16_kernel(
    const float* __restrict__ in, __bf16* __restrict__ out, size_t n)
{
    size_t i = (size_t)blockIdx.x * blockDim.x + threadIdx.x;
    size_t stride = (size_t)gridDim.x * blockDim.x;
    for (; i < n; i += stride) out[i] = (__bf16)in[i];
}

// ---------------------------------------------------------------- GEMM
// C[m,n] = sum_k A[m,k] * W[n,k] + bias[n]      (A,W,C bf16; acc f32)
// Block: 128 thr = 4 waves, 32 rows x 256 cols. A slab staged in LDS by TDM.
__global__ __launch_bounds__(128) void gemm_bf16_kernel(
    const __bf16* __restrict__ A, const __bf16* __restrict__ W,
    const float* __restrict__ bias, __bf16* __restrict__ C)
{
    __shared__ unsigned int ldsA[32 * LDS_ROW_DW];   // 66560 B

    const int lane = threadIdx.x & 31;
    const int wave = threadIdx.x >> 5;
    const int half = lane >> 4;     // 0: lanes 0-15, 1: lanes 16-31
    const int sub  = lane & 15;
    const int m0 = blockIdx.x * 32;
    const int n0 = blockIdx.y * 256 + wave * 64;

    // ---- TDM: one tensor_load_to_lds moves 32 rows x 2048B into LDS (padded) ----
    if (wave == 0) {
        const unsigned long long ga =
            (unsigned long long)(uintptr_t)(A + (size_t)m0 * D_DIM);
        const unsigned int la = (unsigned int)(uintptr_t)&ldsA[0]; // low 32b = LDS offset

        u32x4 g0;
        g0[0] = 1u;                                          // count=1 (valid), user mode
        g0[1] = la;                                          // lds_addr (bytes)
        g0[2] = (unsigned int)(ga & 0xFFFFFFFFull);          // global_addr[31:0]
        g0[3] = (unsigned int)((ga >> 32) & 0x01FFFFFFull)   // global_addr[56:32]
              | (2u << 30);                                  // type = 2 ("image")

        i32x8 g1;
        g1[0] = (2 << 16)        // data_size = 4B (dwords)
              | (1 << 20)        // pad_enable
              | (7 << 22)        // pad_interval code 7 = 256 dwords
              | (3 << 25);       // pad_amount  code 3 = 4 dwords
        g1[1] = (512 << 16);     // tensor_dim0[15:0] = 512 dw (row length)
        g1[2] = (32 << 16);      // tensor_dim0[31:16]=0 | tensor_dim1[15:0] = 32 rows
        g1[3] = (512 << 16);     // tensor_dim1[31:16]=0 | tile_dim0 = 512 dw
        g1[4] = 32;              // tile_dim1 = 32, tile_dim2 = 0
        g1[5] = 512;             // tensor_dim0_stride[31:0] = 512 dw
        g1[6] = 0;               // stride hi / dim1_stride lo
        g1[7] = 0;

        i32x4 gz = {0, 0, 0, 0};
#if defined(__clang_major__) && (__clang_major__ >= 23)
        i32x8 gz8 = {0, 0, 0, 0, 0, 0, 0, 0};
        __builtin_amdgcn_tensor_load_to_lds(g0, g1, gz, gz, gz8, 0);
#else
        __builtin_amdgcn_tensor_load_to_lds(g0, g1, gz, gz, 0);
#endif
        __builtin_amdgcn_s_wait_tensorcnt(0);
    }
    __syncthreads();

    v8f acc[2][4] = {};

    for (int k0 = 0; k0 < D_DIM; k0 += 32) {
        BfVec a[2], b[4];
        // A 16x32 layout from LDS: lanes<16 hold K {0..7,16..23}, lanes>=16 {8..15,24..31}
        #pragma unroll
        for (int ri = 0; ri < 2; ri++) {
            const int r   = ri * 16 + sub;
            const int kh0 = k0 + half * 8;        // in halfs
            const int kh1 = kh0 + 16;
            const int d0  = r * LDS_ROW_DW + (kh0 >> 1) + ((kh0 >= 512) ? 4 : 0);
            const int d1  = r * LDS_ROW_DW + (kh1 >> 1) + ((kh1 >= 512) ? 4 : 0);
            a[ri].q[0] = *(const uint4*)&ldsA[d0];
            a[ri].q[1] = *(const uint4*)&ldsA[d1];
        }
        // B 32x16 layout from global: lane j = column n0+ci*16+j, K {0..7,16..23} (lanes<16) / {8..15,24..31}
        #pragma unroll
        for (int ci = 0; ci < 4; ci++) {
            const __bf16* bp = W + (size_t)(n0 + ci * 16 + sub) * D_DIM + k0 + half * 8;
            b[ci].q[0] = *(const uint4*)(bp);
            b[ci].q[1] = *(const uint4*)(bp + 16);
        }
        #pragma unroll
        for (int ri = 0; ri < 2; ri++)
            #pragma unroll
            for (int ci = 0; ci < 4; ci++)
                acc[ri][ci] = __builtin_amdgcn_wmma_f32_16x16x32_bf16(
                    false, a[ri].v, false, b[ci].v, (short)0, acc[ri][ci], false, false);
    }

    // C layout: VGPR r -> row m0+ri*16+r+8*half, col n0+ci*16+sub
    #pragma unroll
    for (int ri = 0; ri < 2; ri++) {
        #pragma unroll
        for (int ci = 0; ci < 4; ci++) {
            const int ncol = n0 + ci * 16 + sub;
            const float bv = bias[ncol];
            #pragma unroll
            for (int r = 0; r < 8; r++) {
                const int row = m0 + ri * 16 + r + half * 8;
                C[(size_t)row * D_DIM + ncol] = (__bf16)(acc[ri][ci][r] + bv);
            }
        }
    }
}

// ---------------------------------------------------------------- attention
// block = (head h, batch*time bt); 4 waves handle 17 tiles of 16 query rows.
__global__ __launch_bounds__(128) void attn_kernel(
    const __bf16* __restrict__ Q, const __bf16* __restrict__ K,
    const __bf16* __restrict__ V, const float* __restrict__ mask,
    float* __restrict__ out)
{
    __shared__ __bf16 lds_p[4][16][32];   // per-wave probs staging (A layout via LDS)

    const int lane = threadIdx.x & 31;
    const int wave = threadIdx.x >> 5;
    const int half = lane >> 4;
    const int sub  = lane & 15;
    const int h  = blockIdx.x;
    const int bt = blockIdx.y;

    { // zero K-pad columns once (cols 16..31 stay zero forever)
        __bf16* p = &lds_p[0][0][0];
        for (int i = threadIdx.x; i < 4 * 16 * 32; i += 128) p[i] = (__bf16)0.f;
    }
    __syncthreads();

    const __bf16* qb = Q + (size_t)(bt * N_IMG) * D_DIM + h * 64;
    const __bf16* kb = K + (size_t)(bt * L_LM) * D_DIM + h * 64;
    const __bf16* vb = V + (size_t)(bt * L_LM) * D_DIM + h * 64;
    const float*  mb = mask + (size_t)bt * N_IMG * L_LM;
    float*        ob = out + (size_t)(bt * N_IMG) * D_DIM + h * 64;

    // B tiles for scores: k^T (K-dim = hd, N = l). lane col = l = sub.
    BfVec bk2[2];
    #pragma unroll
    for (int kk = 0; kk < 2; kk++) {
        if (sub < L_LM) {
            const __bf16* p = kb + (size_t)sub * D_DIM + kk * 32 + half * 8;
            bk2[kk].q[0] = *(const uint4*)(p);
            bk2[kk].q[1] = *(const uint4*)(p + 16);
        } else {
            bk2[kk].q[0] = make_uint4(0, 0, 0, 0);
            bk2[kk].q[1] = make_uint4(0, 0, 0, 0);
        }
    }
    // B tiles for attn: v (K-dim = l padded to 32, N = hd). lane col hd = ci*16+sub.
    BfVec bv4[4];
    #pragma unroll
    for (int ci = 0; ci < 4; ci++) {
        #pragma unroll
        for (int t = 0; t < 16; t++) {
            const int l = half * 16 + t;
            bv4[ci].h[t] = (l < L_LM) ? vb[(size_t)l * D_DIM + ci * 16 + sub] : (__bf16)0.f;
        }
    }

    for (int it = 0; it < 5; it++) {
        const int tile = wave + it * 4;          // 0..19
        const bool active = (tile <= 16);
        const int tt = active ? tile : 16;       // clamp; keep EXEC all-ones
        const int nbase = tt * 16;

        // Q tile, A layout
        BfVec aq[2];
        const int qrow = nbase + sub;
        const bool rvalid = qrow < N_IMG;
        #pragma unroll
        for (int kk = 0; kk < 2; kk++) {
            if (rvalid) {
                const __bf16* p = qb + (size_t)qrow * D_DIM + kk * 32 + half * 8;
                aq[kk].q[0] = *(const uint4*)(p);
                aq[kk].q[1] = *(const uint4*)(p + 16);
            } else {
                aq[kk].q[0] = make_uint4(0, 0, 0, 0);
                aq[kk].q[1] = make_uint4(0, 0, 0, 0);
            }
        }

        v8f s = {};
        s = __builtin_amdgcn_wmma_f32_16x16x32_bf16(false, aq[0].v, false, bk2[0].v,
                                                    (short)0, s, false, false);
        s = __builtin_amdgcn_wmma_f32_16x16x32_bf16(false, aq[1].v, false, bk2[1].v,
                                                    (short)0, s, false, false);

        // scale + mask + softmax over l (16-lane-half butterfly), write probs -> LDS
        const int l = sub;
        #pragma unroll
        for (int r = 0; r < 8; r++) {
            const int n = nbase + r + half * 8;
            float x = s[r] * 0.125f;  // 1/sqrt(64)
            if (l < L_LM) {
                if (n < N_IMG) x += mb[(size_t)n * L_LM + l];
            } else {
                x = -__builtin_inff();
            }
            float mx = x;
            mx = fmaxf(mx, __shfl_xor(mx, 1, 32));
            mx = fmaxf(mx, __shfl_xor(mx, 2, 32));
            mx = fmaxf(mx, __shfl_xor(mx, 4, 32));
            mx = fmaxf(mx, __shfl_xor(mx, 8, 32));
            float e = __expf(x - mx);
            float sm = e;
            sm += __shfl_xor(sm, 1, 32);
            sm += __shfl_xor(sm, 2, 32);
            sm += __shfl_xor(sm, 4, 32);
            sm += __shfl_xor(sm, 8, 32);
            lds_p[wave][r + half * 8][l] = (__bf16)(e / sm);
        }
        // intra-wave LDS visibility (each wave owns its slab)
        asm volatile("s_wait_dscnt 0" ::: "memory");

        // read probs in A layout (K pad cols are pre-zeroed)
        BfVec ap;
        {
            const __bf16* p = &lds_p[wave][sub][half * 16];
            ap.q[0] = *(const uint4*)(p);
            ap.q[1] = *(const uint4*)(p + 8);
        }

        v8f c[4];
        #pragma unroll
        for (int ci = 0; ci < 4; ci++) {
            v8f z = {};
            c[ci] = __builtin_amdgcn_wmma_f32_16x16x32_bf16(false, ap.v, false, bv4[ci].v,
                                                            (short)0, z, false, false);
        }

        #pragma unroll
        for (int ci = 0; ci < 4; ci++) {
            #pragma unroll
            for (int r = 0; r < 8; r++) {
                const int n = nbase + r + half * 8;
                if (active && n < N_IMG)
                    ob[(size_t)n * D_DIM + ci * 16 + sub] = c[ci][r];
            }
        }
    }
}

// ---------------------------------------------------------------- residual + LN
__global__ __launch_bounds__(256) void ln_kernel(
    const float* __restrict__ img, float* __restrict__ io,
    const float* __restrict__ gamma, const float* __restrict__ beta)
{
    const size_t row = blockIdx.x;
    const int t = threadIdx.x;
    const float4* xi = (const float4*)(img + row * D_DIM);
    float4*       xo = (float4*)(io + row * D_DIM);

    float4 a = xo[t];  // attn
    float4 b = xi[t];  // residual
    float4 x = make_float4(a.x + b.x, a.y + b.y, a.z + b.z, a.w + b.w);

    float s1 = x.x + x.y + x.z + x.w;
    float s2 = x.x * x.x + x.y * x.y + x.z * x.z + x.w * x.w;
    #pragma unroll
    for (int m = 1; m < 32; m <<= 1) {
        s1 += __shfl_xor(s1, m, 32);
        s2 += __shfl_xor(s2, m, 32);
    }
    __shared__ float ps1[8], ps2[8];
    const int wv = t >> 5, ln = t & 31;
    if (ln == 0) { ps1[wv] = s1; ps2[wv] = s2; }
    __syncthreads();
    if (wv == 0) {
        float a1 = (ln < 8) ? ps1[ln] : 0.f;
        float a2 = (ln < 8) ? ps2[ln] : 0.f;
        #pragma unroll
        for (int m = 1; m < 8; m <<= 1) {
            a1 += __shfl_xor(a1, m, 32);
            a2 += __shfl_xor(a2, m, 32);
        }
        if (ln == 0) { ps1[0] = a1; ps2[0] = a2; }
    }
    __syncthreads();
    const float mu  = ps1[0] * (1.f / D_DIM);
    const float var = ps2[0] * (1.f / D_DIM) - mu * mu;
    const float inv = rsqrtf(var + 1e-5f);

    float4 g  = ((const float4*)gamma)[t];
    float4 be = ((const float4*)beta)[t];
    float4 y = make_float4((x.x - mu) * inv * g.x + be.x,
                           (x.y - mu) * inv * g.y + be.y,
                           (x.z - mu) * inv * g.z + be.z,
                           (x.w - mu) * inv * g.w + be.w);
    xo[t] = y;
}

// ---------------------------------------------------------------- launch
extern "C" void kernel_launch(void* const* d_in, const int* in_sizes, int n_in,
                              void* d_out, int out_size, void* d_ws, size_t ws_size,
                              hipStream_t stream) {
    const float* img   = (const float*)d_in[0];
    const float* lm    = (const float*)d_in[1];
    const float* mask  = (const float*)d_in[2];
    const float* Wq    = (const float*)d_in[3];
    const float* bq    = (const float*)d_in[4];
    const float* Wk    = (const float*)d_in[5];
    const float* bk    = (const float*)d_in[6];
    const float* Wv    = (const float*)d_in[7];
    const float* bv    = (const float*)d_in[8];
    const float* gamma = (const float*)d_in[9];
    const float* beta  = (const float*)d_in[10];
    float* out = (float*)d_out;

    char* ws = (char*)d_ws;
    size_t off = 0;
    auto take = [&](size_t bytes) -> void* {
        void* p = ws + off;
        off += (bytes + 255) & ~(size_t)255;
        return p;
    };
    __bf16* img_bf = (__bf16*)take((size_t)ROWS_Q * D_DIM * 2);
    __bf16* lm_bf  = (__bf16*)take((size_t)ROWS_KV * D_DIM * 2);
    __bf16* wq_bf  = (__bf16*)take((size_t)D_DIM * D_DIM * 2);
    __bf16* wk_bf  = (__bf16*)take((size_t)D_DIM * D_DIM * 2);
    __bf16* wv_bf  = (__bf16*)take((size_t)D_DIM * D_DIM * 2);
    __bf16* q_bf   = (__bf16*)take((size_t)ROWS_Q * D_DIM * 2);
    __bf16* k_bf   = (__bf16*)take((size_t)ROWS_KV * D_DIM * 2);
    __bf16* v_bf   = (__bf16*)take((size_t)ROWS_KV * D_DIM * 2);

    auto conv = [&](const float* src, __bf16* dst, size_t n) {
        int blocks = (int)((n + 255) / 256);
        f32_to_bf16_kernel<<<blocks, 256, 0, stream>>>(src, dst, n);
    };
    conv(img, img_bf, (size_t)ROWS_Q * D_DIM);
    conv(lm,  lm_bf,  (size_t)ROWS_KV * D_DIM);
    conv(Wq,  wq_bf,  (size_t)D_DIM * D_DIM);
    conv(Wk,  wk_bf,  (size_t)D_DIM * D_DIM);
    conv(Wv,  wv_bf,  (size_t)D_DIM * D_DIM);

    gemm_bf16_kernel<<<dim3(ROWS_Q / 32, 4), 128, 0, stream>>>(img_bf, wq_bf, bq, q_bf);
    gemm_bf16_kernel<<<dim3(ROWS_KV / 32, 4), 128, 0, stream>>>(lm_bf, wk_bf, bk, k_bf);
    gemm_bf16_kernel<<<dim3(ROWS_KV / 32, 4), 128, 0, stream>>>(lm_bf, wv_bf, bv, v_bf);

    attn_kernel<<<dim3(HEADS, BT_DIM), 128, 0, stream>>>(q_bf, k_bf, v_bf, mask, out);

    ln_kernel<<<ROWS_Q, 256, 0, stream>>>(img, out, gamma, beta);
}
